// CausalSelfAttention_35673998360585
// MI455X (gfx1250) — matrix operand order repaired and checked by reading, not compiled, not run on hardware
//
#include <hip/hip_runtime.h>
#include <hip/hip_bf16.h>

// ---------------------------------------------------------------------------
// CDNA5 (gfx1250) causal self-attention: bf16 WMMA GEMMs + fp32 flash softmax
//   B=8, T=1024, C=768, H=12, d_k=64
// Roofline: ~115 MB HBM traffic (~5us @23.3TB/s), everything L2-resident;
// matrix work must ride v_wmma_f32_16x16x32_bf16, softmax stays fp32 (exp2).
// ---------------------------------------------------------------------------

typedef unsigned short u16;
typedef __bf16        v16bf __attribute__((ext_vector_type(16)));
typedef u16           v16u  __attribute__((ext_vector_type(16)));
typedef float         v8f   __attribute__((ext_vector_type(8)));
typedef unsigned int  u32x4 __attribute__((ext_vector_type(4)));
typedef int           i32x4 __attribute__((ext_vector_type(4)));
typedef int           i32x8 __attribute__((ext_vector_type(8)));

#if defined(__has_builtin)
#if __has_builtin(__builtin_amdgcn_tensor_load_to_lds) && \
    __has_builtin(__builtin_amdgcn_s_wait_tensorcnt)
#define HAVE_TDM 1
#endif
#endif

#define DEV static __device__ __forceinline__

DEV u16 f32_to_bf16(float f) {
  unsigned u = __builtin_bit_cast(unsigned, f);
  u += 0x7fffu + ((u >> 16) & 1u);           // round-to-nearest-even
  return (u16)(u >> 16);
}

DEV v8f vzero8() {
  v8f r;
#pragma unroll
  for (int i = 0; i < 8; ++i) r[i] = 0.0f;
  return r;
}

// D = A*B + C ; bf16 in, fp32 accumulate (v_wmma_f32_16x16x32_bf16)
DEV v8f wmma_bf16(v16bf a, v16bf b, v8f c) {
  return __builtin_amdgcn_wmma_f32_16x16x32_bf16(
      /*neg_a=*/false, a, /*neg_b=*/false, b,
      /*c_mod=*/(short)0, c, /*reuse_a=*/false, /*reuse_b=*/false);
}

// A fragment (16x32, MxK): lane = hi*16+nl holds row nl,
//   slots 0..15 -> K = hi*16 + j (K contiguous within lane)
DEV v16bf load_afrag(const u16* p, int stride, int lane) {
  const int nl = lane & 15, hi = lane >> 4;
  const u16* r = p + (size_t)nl * stride;
  v16u t;
#pragma unroll
  for (int j = 0; j < 16; ++j) {
    t[j] = r[hi * 16 + j];
  }
  return __builtin_bit_cast(v16bf, t);
}

// B fragment (32x16, KxN): lane holds column n = lane&15,
//   lanes 0-15: K=0..15, lanes 16-31: K=16..31 (K contiguous within lane).
// Memory must be n-major with K fast: element(n,k) = p[n*nstride + k].
DEV v16bf load_bfrag(const u16* p, int nstride, int lane) {
  const int nl = lane & 15, hi = lane >> 4;
  const u16* r = p + (size_t)nl * nstride + hi * 16;
  v16u t;
#pragma unroll
  for (int j = 0; j < 16; ++j) t[j] = r[j];
  return __builtin_bit_cast(v16bf, t);
}

#ifdef HAVE_TDM
// Issue a TDM 2-D tile load (bf16 elements) into LDS. Descriptor per
// CDNA5 ISA: group0 = {flags, lds_addr, global_addr,
// type=2}, group1 = {data_size=2B, tensor dims/strides, tile dims}.
// Caller must be wave-uniform; completion via s_wait_tensorcnt.
DEV void tdm_load_tile_bf16(u16* lds_dst, const u16* gsrc,
                            unsigned tensor_d0, unsigned tensor_d1,
                            unsigned tile_d0, unsigned tile_d1,
                            unsigned stride0) {
  const unsigned long long ga = (unsigned long long)(uintptr_t)gsrc;
  u32x4 g0;
  g0[0] = 1u;                                   // count=1, user mode
  g0[1] = (unsigned)(uintptr_t)lds_dst;         // lds_addr (low 32b = LDS off)
  g0[2] = (unsigned)ga;                         // global_addr[31:0]
  g0[3] = (unsigned)(ga >> 32) | (2u << 30);    // global_addr[56:32] | type=2
  i32x8 g1;
  g1[0] = (int)(1u << 16);                      // data_size=1 (2 bytes)
  g1[1] = (int)(tensor_d0 << 16);               // tensor_dim0[15:0] @ bits 63:48
  g1[2] = (int)((tensor_d0 >> 16) | (tensor_d1 << 16));  // td0 hi | td1 lo
  g1[3] = (int)((tensor_d1 >> 16) | (tile_d0 << 16));    // td1 hi | tile_dim0
  g1[4] = (int)(tile_d1 & 0xffffu);             // tile_dim1 | tile_dim2=0
  g1[5] = (int)stride0;                         // tensor_dim0_stride[31:0]
  g1[6] = 0;                                    // stride0 hi | stride1 lo
  g1[7] = 0;                                    // stride1 hi
  i32x4 gz4 = {0, 0, 0, 0};                     // 2-D tensor: groups 2/3 unused
  i32x8 gz8 = {0, 0, 0, 0, 0, 0, 0, 0};         // trailing group (clang-23 arity)
  __builtin_amdgcn_tensor_load_to_lds(g0, g1, gz4, gz4, gz8, 0);
}
#endif

// ---------------------------------------------------------------------------
// Shared 128x128 GEMM tile engine: 256 threads = 8 waves (4 M-waves x 2
// N-waves); each wave computes 32x64 = 2x4 WMMA accumulators (8 WMMA/stage).
// A: [M x K] (fp32 or bf16), W: fp32 [K x N] row-major. LDS-staged as bf16.
// bf16 A tiles are staged by the Tensor Data Mover when available.
// ---------------------------------------------------------------------------
template <bool AF32>
DEV void gemm_tile_128x128(const void* Aptr, size_t lda, size_t Mrows,
                           const float* __restrict__ W, int ldw, int K,
                           size_t m0, size_t n0,
                           u16* lA /*[128*32]*/, u16* lB /*[128*32] n-major*/,
                           v8f acc[2][4]) {
  const int tid  = threadIdx.x;
  const int lane = tid & 31, w = tid >> 5;
  const int wm = w & 3, wn = w >> 2;

  for (int kk = 0; kk < K; kk += 32) {
    // ---- stage A tile 128x32 (row-major bf16) ----
    if (AF32) {
      const int row = tid >> 1, ch = (tid & 1) * 16;
      const float* src = (const float*)Aptr + (m0 + row) * lda + kk + ch;
#pragma unroll
      for (int j = 0; j < 16; ++j) lA[row * 32 + ch + j] = f32_to_bf16(src[j]);
    } else {
#ifdef HAVE_TDM
      if (w == 0) {  // wave-uniform: one TDM DMA for the whole 8KB tile
        tdm_load_tile_bf16(lA, (const u16*)Aptr + m0 * lda + kk,
                           (unsigned)lda, (unsigned)Mrows,
                           /*tile_d0=*/32u, /*tile_d1=*/128u, (unsigned)lda);
        __builtin_amdgcn_s_wait_tensorcnt(0);
      }
#else
      const int row = tid >> 1, ch = (tid & 1) * 16;
      const u16* src = (const u16*)Aptr + (m0 + row) * lda + kk + ch;
#pragma unroll
      for (int j = 0; j < 16; ++j) lA[row * 32 + ch + j] = src[j];
#endif
    }
    // ---- stage B tile 32(k) x 128(n) -> n-major lB[n][k], fp32->bf16 ----
    {
      const int base = tid * 16;
      const int k = base >> 7;        // 0..31
      const int n = base & 127;       // multiple of 16
      const float* src = W + (size_t)(kk + k) * ldw + n0 + n;
      if (kk + 32 < K)
        __builtin_prefetch(src + (size_t)32 * ldw, 0, 1);  // next K-slab
#pragma unroll
      for (int j = 0; j < 16; ++j) lB[(n + j) * 32 + k] = f32_to_bf16(src[j]);
    }
    __syncthreads();

    const v16bf a0 = load_afrag(lA + (wm * 32 + 0)  * 32, 32, lane);
    const v16bf a1 = load_afrag(lA + (wm * 32 + 16) * 32, 32, lane);
#pragma unroll
    for (int ni = 0; ni < 4; ++ni) {
      const v16bf b = load_bfrag(lB + (wn * 64 + ni * 16) * 32, 32, lane);
      acc[0][ni] = wmma_bf16(a0, b, acc[0][ni]);
      acc[1][ni] = wmma_bf16(a1, b, acc[1][ni]);
    }
    __syncthreads();
  }
}

// ---------------------------------------------------------------------------
// Kernel 1: qkv = x @ W_qkv + b_qkv, scatter into
//   Q [B,H,T,d] bf16 ; K [B,H,T,d] bf16 ; V transposed [B,H,d,T] bf16
// M=8192, K=768, N=2304. Grid (18, 64).
// ---------------------------------------------------------------------------
__global__ __launch_bounds__(256) void qkv_gemm_kernel(
    const float* __restrict__ x, const float* __restrict__ Wqkv,
    const float* __restrict__ bqkv,
    u16* __restrict__ Qb, u16* __restrict__ Kb, u16* __restrict__ Vt) {
  __shared__ u16 lA[128 * 32];
  __shared__ u16 lB[128 * 32];
  v8f acc[2][4] = {{vzero8(), vzero8(), vzero8(), vzero8()},
                   {vzero8(), vzero8(), vzero8(), vzero8()}};

  const size_t m0 = (size_t)blockIdx.y * 128;
  const size_t n0 = (size_t)blockIdx.x * 128;
  gemm_tile_128x128<true>(x, 768, 8192, Wqkv, 2304, 768, m0, n0, lA, lB, acc);

  const int lane = threadIdx.x & 31, w = threadIdx.x >> 5;
  const int wm = w & 3, wn = w >> 2;
  const int hi = lane >> 4, nl = lane & 15;
#pragma unroll
  for (int mi = 0; mi < 2; ++mi)
#pragma unroll
    for (int ni = 0; ni < 4; ++ni)
#pragma unroll
      for (int v = 0; v < 8; ++v) {
        const int gm = (int)m0 + wm * 32 + mi * 16 + hi * 8 + v;
        const int gn = (int)n0 + wn * 64 + ni * 16 + nl;
        const u16 bv = f32_to_bf16(acc[mi][ni][v] + bqkv[gn]);
        const int sel = gn / 768, c = gn % 768;
        const int h = c >> 6, d = c & 63;
        const int bb = gm >> 10, t = gm & 1023;
        const size_t bh = (size_t)bb * 12 + h;
        if (sel == 0)      Qb[(bh * 1024 + t) * 64 + d] = bv;
        else if (sel == 1) Kb[(bh * 1024 + t) * 64 + d] = bv;
        else               Vt[bh * 65536 + (size_t)d * 1024 + t] = bv;
      }
}

// ---------------------------------------------------------------------------
// Kernel 2: causal flash attention, one wave per 16 query rows.
//   S = Q*K^T (4 WMMA / 32-key block), online softmax (exp2 domain),
//   P bounced through per-wave LDS (D-layout -> A-layout), O += P*V (4 WMMA).
// Grid: (B*H, T/128), block 256 = 8 waves.
// ---------------------------------------------------------------------------
__global__ __launch_bounds__(256) void attn_kernel(
    const u16* __restrict__ Qb, const u16* __restrict__ Kb,
    const u16* __restrict__ Vt, u16* __restrict__ attn) {
  __shared__ u16 Plds[8 * 16 * 32];

  const int lane = threadIdx.x & 31, w = threadIdx.x >> 5;
  const int hi = lane >> 4, nl = lane & 15;
  const int bh = blockIdx.x;                 // 0..95
  const int q0 = blockIdx.y * 128 + w * 16;  // wave's query-row base

  const u16* Q = Qb + (size_t)bh * 65536;    // [T,64]
  const u16* K = Kb + (size_t)bh * 65536;    // [T,64]
  const u16* V = Vt + (size_t)bh * 65536;    // [64,T] (transposed)
  u16* P = Plds + w * 512;                   // per-wave 16x32 bf16 tile

  // Q fragments for d=0..31 and d=32..63 (reused across all key blocks)
  const v16bf qa0 = load_afrag(Q + (size_t)q0 * 64,      64, lane);
  const v16bf qa1 = load_afrag(Q + (size_t)q0 * 64 + 32, 64, lane);

  v8f o[4] = {vzero8(), vzero8(), vzero8(), vzero8()};
  float rm[8], rs[8];
#pragma unroll
  for (int v = 0; v < 8; ++v) { rm[v] = -__builtin_inff(); rs[v] = 0.0f; }

  const float SC = 0.125f * 1.44269504088896340736f;  // d_k^-0.5 * log2(e)

  for (int key0 = 0; key0 <= q0 + 15; key0 += 32) {   // causal loop bound
    // K^T B-fragments: element(n=key, k=d) = K[key][d] -> 32B contiguous/lane
    const v16bf k00 = load_bfrag(K + (size_t)key0 * 64,             64, lane);
    const v16bf k01 = load_bfrag(K + (size_t)key0 * 64 + 32,        64, lane);
    const v16bf k10 = load_bfrag(K + (size_t)(key0 + 16) * 64,      64, lane);
    const v16bf k11 = load_bfrag(K + (size_t)(key0 + 16) * 64 + 32, 64, lane);
    if (key0 + 32 <= q0 + 15) {  // prefetch next key block (K and V streams)
      __builtin_prefetch(K + (size_t)(key0 + 32 + nl) * 64, 0, 1);
      __builtin_prefetch(V + (size_t)nl * 1024 + key0 + 32, 0, 1);
    }

    v8f s0 = vzero8(), s1 = vzero8();
    s0 = wmma_bf16(qa0, k00, s0);
    s0 = wmma_bf16(qa1, k01, s0);
    s1 = wmma_bf16(qa0, k10, s1);
    s1 = wmma_bf16(qa1, k11, s1);

#pragma unroll
    for (int v = 0; v < 8; ++v) {
      const int row = q0 + hi * 8 + v;
      float a = s0[v] * SC; if (key0 + nl > row)      a = -__builtin_inff();
      float b = s1[v] * SC; if (key0 + 16 + nl > row) b = -__builtin_inff();
      // row max across the 16-lane half (xor<16 stays within half)
      float m = fmaxf(a, b);
      m = fmaxf(m, __shfl_xor(m, 1, 32));
      m = fmaxf(m, __shfl_xor(m, 2, 32));
      m = fmaxf(m, __shfl_xor(m, 4, 32));
      m = fmaxf(m, __shfl_xor(m, 8, 32));
      const float mnew  = fmaxf(rm[v], m);
      const float alpha = __builtin_amdgcn_exp2f(rm[v] - mnew);
      const float p0    = __builtin_amdgcn_exp2f(a - mnew);
      const float p1    = __builtin_amdgcn_exp2f(b - mnew);
      float sum = p0 + p1;
      sum += __shfl_xor(sum, 1, 32);
      sum += __shfl_xor(sum, 2, 32);
      sum += __shfl_xor(sum, 4, 32);
      sum += __shfl_xor(sum, 8, 32);
      rs[v] = rs[v] * alpha + sum;
      rm[v] = mnew;
#pragma unroll
      for (int t = 0; t < 4; ++t) o[t][v] *= alpha;
      // store P in D-layout: row = hi*8+v, col = nl (+16)
      P[(hi * 8 + v) * 32 + nl]      = f32_to_bf16(p0);
      P[(hi * 8 + v) * 32 + 16 + nl] = f32_to_bf16(p1);
    }
    asm volatile("s_wait_dscnt 0" ::: "memory");  // LDS RAW before A-frag read

    const v16bf pa = load_afrag(P, 32, lane);     // P as A-fragment (16x32)
    // V B-fragments: element(n=d, k=key) = Vt[d][key] -> 32B contiguous/lane
    const v16bf vb0 = load_bfrag(V + 0 * 16384 + key0, 1024, lane);
    const v16bf vb1 = load_bfrag(V + 1 * 16384 + key0, 1024, lane);
    const v16bf vb2 = load_bfrag(V + 2 * 16384 + key0, 1024, lane);
    const v16bf vb3 = load_bfrag(V + 3 * 16384 + key0, 1024, lane);
    o[0] = wmma_bf16(pa, vb0, o[0]);
    o[1] = wmma_bf16(pa, vb1, o[1]);
    o[2] = wmma_bf16(pa, vb2, o[2]);
    o[3] = wmma_bf16(pa, vb3, o[3]);
  }

  // normalize and write attn [B,T,C] bf16, c = h*64 + d
  const int bb = bh / 12, h = bh % 12;
#pragma unroll
  for (int v = 0; v < 8; ++v) {
    const float inv = 1.0f / rs[v];
    const int t = q0 + hi * 8 + v;
    const size_t base = ((size_t)bb * 1024 + t) * 768 + h * 64;
#pragma unroll
    for (int dc = 0; dc < 4; ++dc)
      attn[base + dc * 16 + nl] = f32_to_bf16(o[dc][v] * inv);
  }
}

// ---------------------------------------------------------------------------
// Kernel 3: out = attn(bf16) @ W_out + b_out -> fp32. M=8192, K=768, N=768.
// Grid (6, 64). A tiles staged via Tensor Data Mover when available.
// ---------------------------------------------------------------------------
__global__ __launch_bounds__(256) void out_gemm_kernel(
    const u16* __restrict__ attn, const float* __restrict__ Wout,
    const float* __restrict__ bout, float* __restrict__ out) {
  __shared__ u16 lA[128 * 32];
  __shared__ u16 lB[128 * 32];
  v8f acc[2][4] = {{vzero8(), vzero8(), vzero8(), vzero8()},
                   {vzero8(), vzero8(), vzero8(), vzero8()}};

  const size_t m0 = (size_t)blockIdx.y * 128;
  const size_t n0 = (size_t)blockIdx.x * 128;
  gemm_tile_128x128<false>(attn, 768, 8192, Wout, 768, 768, m0, n0, lA, lB, acc);

  const int lane = threadIdx.x & 31, w = threadIdx.x >> 5;
  const int wm = w & 3, wn = w >> 2;
  const int hi = lane >> 4, nl = lane & 15;
#pragma unroll
  for (int mi = 0; mi < 2; ++mi)
#pragma unroll
    for (int ni = 0; ni < 4; ++ni)
#pragma unroll
      for (int v = 0; v < 8; ++v) {
        const int gm = (int)m0 + wm * 32 + mi * 16 + hi * 8 + v;
        const int gn = (int)n0 + wn * 64 + ni * 16 + nl;
        out[(size_t)gm * 768 + gn] = acc[mi][ni][v] + bout[gn];
      }
}

// ---------------------------------------------------------------------------
extern "C" void kernel_launch(void* const* d_in, const int* in_sizes, int n_in,
                              void* d_out, int out_size, void* d_ws, size_t ws_size,
                              hipStream_t stream) {
  (void)in_sizes; (void)n_in; (void)out_size; (void)ws_size;
  const float* x    = (const float*)d_in[0];  // [8,1024,768]
  const float* Wqkv = (const float*)d_in[1];  // [768,2304]
  const float* bqkv = (const float*)d_in[2];  // [2304]
  const float* Wout = (const float*)d_in[3];  // [768,768]
  const float* bout = (const float*)d_in[4];  // [768]
  float* out = (float*)d_out;                 // [8,1024,768]

  const size_t NE = (size_t)8 * 12 * 1024 * 64;  // 6,291,456 elems per tensor
  u16* Qb    = (u16*)d_ws;        // [B,H,T,d] bf16
  u16* Kb    = Qb + NE;           // [B,H,T,d] bf16
  u16* Vt    = Kb + NE;           // [B,H,d,T] bf16 (transposed)
  u16* attnb = Vt + NE;           // [B,T,C]   bf16

  dim3 blk(256);
  qkv_gemm_kernel<<<dim3(18, 64), blk, 0, stream>>>(x, Wqkv, bqkv, Qb, Kb, Vt);
  attn_kernel    <<<dim3(96, 8),  blk, 0, stream>>>(Qb, Kb, Vt, attnb);
  out_gemm_kernel<<<dim3(6, 64),  blk, 0, stream>>>(attnb, Wout, bout, out);
}
